// PCGTConvLayer_66821101191697
// MI455X (gfx1250) — compile-verified, matrix operand order checked by reading, not run on hardware
//
#include <hip/hip_runtime.h>
#include <hip/hip_bf16.h>
#include <math.h>

typedef _Float16 h_t;
typedef __attribute__((ext_vector_type(16))) _Float16 v16h;
typedef __attribute__((ext_vector_type(8)))  _Float16 v8h;
typedef __attribute__((ext_vector_type(8)))  float    v8f;
typedef __attribute__((ext_vector_type(4)))  float    v4f;

#define N_NODES 32768
#define IN_CH   256
#define DHEAD   128
#define NHEADS  4
#define MREPS   4
#define KPART   128
#define SPART   256
#define HD      512          // NHEADS*DHEAD
#define NREPS   512          // KPART*MREPS
#define SCALE_INV 0.088388347648318447f   // 1/sqrt(128)

// ---------------------------------------------------------------------------
// WMMA helpers (CDNA5 wave32, 16x16x32 f16 -> f32)
// ---------------------------------------------------------------------------
static __device__ __forceinline__ v8f wmma_f16(v16h a, v16h b, v8f c) {
  // args: (neg_a, A, neg_b, B, c_mod, C, reuse_a, reuse_b)
  return __builtin_amdgcn_wmma_f32_16x16x32_f16(false, a, false, b, (short)0, c,
                                                false, false);
}

// Load a 16-half fragment for lane-group g from a row-major f16 row pointer.
// Lane-group 0 takes k runs {0..7, 16..23}; group 1 takes {8..15, 24..31}.
static __device__ __forceinline__ v16h load_frag16(const h_t* rowk0, int g) {
  v8h lo = *(const v8h*)(rowk0 + g * 8);
  v8h hi = *(const v8h*)(rowk0 + 16 + g * 8);
  v16h r;
#pragma unroll
  for (int j = 0; j < 8; ++j) { r[j] = lo[j]; r[8 + j] = hi[j]; }
  return r;
}

// Same fragment but sourced from f32 (converted on the fly).
static __device__ __forceinline__ v16h load_frag_f32(const float* rowk0, int g) {
  const float* p0 = rowk0 + g * 8;
  const float* p1 = rowk0 + 16 + g * 8;
  v4f a0 = *(const v4f*)p0, a1 = *(const v4f*)(p0 + 4);
  v4f b0 = *(const v4f*)p1, b1 = *(const v4f*)(p1 + 4);
  v16h r;
#pragma unroll
  for (int j = 0; j < 4; ++j) {
    r[j] = (h_t)a0[j]; r[4 + j] = (h_t)a1[j];
    r[8 + j] = (h_t)b0[j]; r[12 + j] = (h_t)b1[j];
  }
  return r;
}

// CDNA5 async global->LDS copy (16B per lane), tracked by ASYNCcnt.
static __device__ __forceinline__ void async_b128(unsigned lds_off,
                                                  const h_t* gptr) {
  asm volatile("global_load_async_to_lds_b128 %0, %1, off"
               :
               : "v"(lds_off), "v"((unsigned long long)(uintptr_t)gptr)
               : "memory");
}

// 16-lane-half reductions (rows of a C-fragment live inside one half in wave32)
static __device__ __forceinline__ float redmax16(float v) {
  v = fmaxf(v, __shfl_xor(v, 1));
  v = fmaxf(v, __shfl_xor(v, 2));
  v = fmaxf(v, __shfl_xor(v, 4));
  v = fmaxf(v, __shfl_xor(v, 8));
  return v;
}
static __device__ __forceinline__ float redsum16(float v) {
  v += __shfl_xor(v, 1);
  v += __shfl_xor(v, 2);
  v += __shfl_xor(v, 4);
  v += __shfl_xor(v, 8);
  return v;
}

// ---------------------------------------------------------------------------
// Flash-attention inner loop (global-streaming variant, used for the small
// L2-resident reps table in cross-attention). One wave = 16 query rows.
// ---------------------------------------------------------------------------
static __device__ __forceinline__ void flash_rows_global(
    const h_t* __restrict__ Kmat, const h_t* __restrict__ Vmat, int head,
    int nkeys, const v16h* qfrag, h_t* pbuf /*wave-private [16*32]*/,
    v8f* oacc, float* l) {
  const int lane = threadIdx.x & 31;
  const int li = lane & 15;
  const int g = lane >> 4;
  const int hoff = head * DHEAD;
  float m[8];
#pragma unroll
  for (int i = 0; i < 8; ++i) { m[i] = -__builtin_inff(); l[i] = 0.f; }

  for (int jb = 0; jb < nkeys; jb += 32) {
    // ---- logits tile: 16 queries x 32 keys, contract over D=128 ----
    const h_t* ka = Kmat + (size_t)(jb + li) * HD + hoff;
    const h_t* kb = Kmat + (size_t)(jb + 16 + li) * HD + hoff;
    v8f s0 = {0.f, 0.f, 0.f, 0.f, 0.f, 0.f, 0.f, 0.f};
    v8f s1 = {0.f, 0.f, 0.f, 0.f, 0.f, 0.f, 0.f, 0.f};
#pragma unroll
    for (int kc = 0; kc < 4; ++kc) {
      v16h bk0 = load_frag16(ka + kc * 32, g);
      v16h bk1 = load_frag16(kb + kc * 32, g);
      s0 = wmma_f16(qfrag[kc], bk0, s0);
      s1 = wmma_f16(qfrag[kc], bk1, s1);
    }
    // ---- online softmax update ----
    float corr[8];
#pragma unroll
    for (int i = 0; i < 8; ++i) {
      float a = s0[i] * SCALE_INV, b = s1[i] * SCALE_INV;
      float tm = redmax16(fmaxf(a, b));
      float mn = fmaxf(m[i], tm);
      corr[i] = __expf(m[i] - mn);
      m[i] = mn;
      float p0 = __expf(a - mn), p1 = __expf(b - mn);
      s0[i] = p0; s1[i] = p1;
      l[i] = l[i] * corr[i] + redsum16(p0 + p1);
    }
#pragma unroll
    for (int t = 0; t < 8; ++t) {
#pragma unroll
      for (int i = 0; i < 8; ++i) oacc[t][i] *= corr[i];
    }
    // ---- C-frag -> A-frag relayout of P through wave-private LDS ----
#pragma unroll
    for (int i = 0; i < 8; ++i) {
      const int row = i + 8 * g;
      pbuf[row * 32 + li] = (h_t)s0[i];
      pbuf[row * 32 + 16 + li] = (h_t)s1[i];
    }
    asm volatile("s_wait_dscnt 0" ::: "memory");
    const v16h pf = load_frag16(pbuf + li * 32, g);
    // ---- accumulate: out += P(16x32) * V(32x128) ----
#pragma unroll
    for (int t = 0; t < 8; ++t) {
      v16h bvf;
#pragma unroll
      for (int j = 0; j < 8; ++j) {
        const int n1 = jb + g * 8 + j;
        const int n2 = jb + 16 + g * 8 + j;
        bvf[j] = Vmat[(size_t)n1 * HD + hoff + t * 16 + li];
        bvf[8 + j] = Vmat[(size_t)n2 * HD + hoff + t * 16 + li];
      }
      oacc[t] = wmma_f16(pf, bvf, oacc[t]);
    }
  }
}

// ---------------------------------------------------------------------------
// Kernel 0: convert weights to transposed f16 [mat][col][k]
// ---------------------------------------------------------------------------
__global__ __launch_bounds__(256) void convert_w_kernel(
    const float* __restrict__ wq, const float* __restrict__ wk,
    const float* __restrict__ wv, h_t* __restrict__ wt16) {
  const int i = blockIdx.x * 256 + threadIdx.x;
  if (i < 3 * IN_CH * HD) {
    const int mat = i / (IN_CH * HD);
    const int rem = i - mat * (IN_CH * HD);
    const int r = rem >> 9;   // k index 0..255
    const int c = rem & 511;  // col index 0..511
    const float* w = (mat == 0) ? wq : ((mat == 1) ? wk : wv);
    wt16[(size_t)mat * (HD * IN_CH) + (size_t)c * IN_CH + r] = (h_t)w[rem];
  }
}

// ---------------------------------------------------------------------------
// Kernel 1: QKV GEMM.  One wave computes a 16x128 tile of one of Q/K/V.
// ---------------------------------------------------------------------------
__global__ __launch_bounds__(32) void qkv_gemm_kernel(
    const float* __restrict__ x, const h_t* __restrict__ wt16,
    const float* __restrict__ bq, const float* __restrict__ bk,
    const float* __restrict__ bv, h_t* __restrict__ Q16,
    h_t* __restrict__ K16, h_t* __restrict__ V16) {
  const int rb = blockIdx.x;   // 0..2047 row blocks of 16
  const int cb = blockIdx.y;   // 0..3 col blocks of 128
  const int mat = blockIdx.z;  // 0=Q 1=K 2=V
  const int lane = threadIdx.x & 31;
  const int li = lane & 15, g = lane >> 4;
  const h_t* wt = wt16 + (size_t)mat * (HD * IN_CH);
  const float* bias = (mat == 0) ? bq : ((mat == 1) ? bk : bv);
  h_t* out = (mat == 0) ? Q16 : ((mat == 1) ? K16 : V16);
  const int r = rb * 16 + li;
  const int c0 = cb * 128;

  v8f acc[8];
#pragma unroll
  for (int t = 0; t < 8; ++t) acc[t] = (v8f){0.f,0.f,0.f,0.f,0.f,0.f,0.f,0.f};

  for (int k0 = 0; k0 < IN_CH; k0 += 32) {
    const v16h af = load_frag_f32(x + (size_t)r * IN_CH + k0, g);
#pragma unroll
    for (int t = 0; t < 8; ++t) {
      const int c = c0 + t * 16 + li;
      const v16h bf = load_frag16(wt + (size_t)c * IN_CH + k0, g);
      acc[t] = wmma_f16(af, bf, acc[t]);
    }
  }
#pragma unroll
  for (int t = 0; t < 8; ++t) {
    const int c = c0 + t * 16 + li;
    const float bval = bias[c];
#pragma unroll
    for (int i = 0; i < 8; ++i) {
      const int row = rb * 16 + i + 8 * g;
      out[(size_t)row * HD + c] = (h_t)(acc[t][i] + bval);
    }
  }
}

// ---------------------------------------------------------------------------
// Kernel 2: per-partition exact softmax attention with async-DMA staging.
// 1 block per partition: 16 waves x 16 query rows, loop over 4 heads.
// Gathered 32x128 K/V tiles are staged global->LDS with
// GLOBAL_LOAD_ASYNC_TO_LDS_B128 (double-buffered, overlapped with WMMA),
// synchronized with s_wait_asynccnt + workgroup barriers.
// ---------------------------------------------------------------------------
__global__ __launch_bounds__(512) void local_attn_kernel(
    const h_t* __restrict__ Q16, const h_t* __restrict__ K16,
    const h_t* __restrict__ V16, const int* __restrict__ pidx,
    float* __restrict__ x_local) {
  __shared__ alignas(16) h_t ktile[2][32][DHEAD];   // 16 KB
  __shared__ alignas(16) h_t vtile[2][32][DHEAD];   // 16 KB
  __shared__ alignas(16) h_t pbuf_s[16][16 * 32];   // 16 KB
  __shared__ int nodes_s[SPART];                    // 1 KB

  const int part = blockIdx.x;
  const int tid = threadIdx.x;
  const int w = tid >> 5;
  const int lane = tid & 31;
  const int li = lane & 15, g = lane >> 4;
  const int* pp = pidx + part * SPART;

  if (tid < SPART) nodes_s[tid] = pp[tid];
  __syncthreads();

  const int qnode = nodes_s[w * 16 + li];
  int onode[8];
#pragma unroll
  for (int i = 0; i < 8; ++i) onode[i] = nodes_s[w * 16 + i + 8 * g];

  // staging role: 512 threads x 16B == one 32x128 f16 tile per instruction
  const int srow = tid >> 4;            // 0..31 key row within tile
  const int schunk = (tid & 15) * 8;    // half-offset within 128-half row

  for (int h = 0; h < NHEADS; ++h) {
    v16h qfrag[4];
    const h_t* qbase = Q16 + (size_t)qnode * HD + h * DHEAD;
#pragma unroll
    for (int kc = 0; kc < 4; ++kc) qfrag[kc] = load_frag16(qbase + kc * 32, g);
    v8f oacc[8];
#pragma unroll
    for (int t = 0; t < 8; ++t) oacc[t] = (v8f){0.f,0.f,0.f,0.f,0.f,0.f,0.f,0.f};
    float m[8], l[8];
#pragma unroll
    for (int i = 0; i < 8; ++i) { m[i] = -__builtin_inff(); l[i] = 0.f; }

    { // prologue: stage tile 0 into buffer 0
      const int node = nodes_s[srow];
      async_b128((unsigned)(uintptr_t)&ktile[0][srow][schunk],
                 K16 + (size_t)node * HD + h * DHEAD + schunk);
      async_b128((unsigned)(uintptr_t)&vtile[0][srow][schunk],
                 V16 + (size_t)node * HD + h * DHEAD + schunk);
    }

    for (int jb = 0; jb < SPART; jb += 32) {
      const int bb = (jb >> 5) & 1;
      if (jb + 32 < SPART) {
        const int node = nodes_s[jb + 32 + srow];
        async_b128((unsigned)(uintptr_t)&ktile[bb ^ 1][srow][schunk],
                   K16 + (size_t)node * HD + h * DHEAD + schunk);
        async_b128((unsigned)(uintptr_t)&vtile[bb ^ 1][srow][schunk],
                   V16 + (size_t)node * HD + h * DHEAD + schunk);
        asm volatile("s_wait_asynccnt 0x2" ::: "memory");  // tile jb landed
      } else {
        asm volatile("s_wait_asynccnt 0x0" ::: "memory");
      }
      __syncthreads();  // all waves' DMA for tile jb visible

      // ---- logits tile 16x32 from LDS K tile ----
      const h_t* ka = &ktile[bb][li][0];
      const h_t* kb2 = &ktile[bb][16 + li][0];
      v8f s0 = {0.f, 0.f, 0.f, 0.f, 0.f, 0.f, 0.f, 0.f};
      v8f s1 = {0.f, 0.f, 0.f, 0.f, 0.f, 0.f, 0.f, 0.f};
#pragma unroll
      for (int kc = 0; kc < 4; ++kc) {
        v16h bk0 = load_frag16(ka + kc * 32, g);
        v16h bk1 = load_frag16(kb2 + kc * 32, g);
        s0 = wmma_f16(qfrag[kc], bk0, s0);
        s1 = wmma_f16(qfrag[kc], bk1, s1);
      }
      // ---- online softmax update ----
      float corr[8];
#pragma unroll
      for (int i = 0; i < 8; ++i) {
        float a = s0[i] * SCALE_INV, b = s1[i] * SCALE_INV;
        float tm = redmax16(fmaxf(a, b));
        float mn = fmaxf(m[i], tm);
        corr[i] = __expf(m[i] - mn);
        m[i] = mn;
        float p0 = __expf(a - mn), p1 = __expf(b - mn);
        s0[i] = p0; s1[i] = p1;
        l[i] = l[i] * corr[i] + redsum16(p0 + p1);
      }
#pragma unroll
      for (int t = 0; t < 8; ++t) {
#pragma unroll
        for (int i = 0; i < 8; ++i) oacc[t][i] *= corr[i];
      }
      // ---- C-frag -> A-frag relayout of P via wave-private LDS ----
#pragma unroll
      for (int i = 0; i < 8; ++i) {
        const int row = i + 8 * g;
        pbuf_s[w][row * 32 + li] = (h_t)s0[i];
        pbuf_s[w][row * 32 + 16 + li] = (h_t)s1[i];
      }
      asm volatile("s_wait_dscnt 0" ::: "memory");
      const v16h pf = load_frag16(&pbuf_s[w][0] + li * 32, g);
      // ---- accumulate out += P(16x32) * Vtile(32x128) from LDS ----
#pragma unroll
      for (int t = 0; t < 8; ++t) {
        v16h bvf;
#pragma unroll
        for (int j = 0; j < 8; ++j) {
          bvf[j] = vtile[bb][g * 8 + j][t * 16 + li];
          bvf[8 + j] = vtile[bb][16 + g * 8 + j][t * 16 + li];
        }
        oacc[t] = wmma_f16(pf, bvf, oacc[t]);
      }
      __syncthreads();  // tile consumed; its buffer may be re-staged
    }

    // ---- accumulate head mean into x_local (rows exclusive per wave) ----
#pragma unroll
    for (int t = 0; t < 8; ++t) {
#pragma unroll
      for (int i = 0; i < 8; ++i) {
        const float v = oacc[t][i] / l[i] * 0.25f;
        float* dst = x_local + (size_t)onode[i] * DHEAD + t * 16 + li;
        if (h == 0) *dst = v; else *dst += v;
      }
    }
  }
}

// ---------------------------------------------------------------------------
// Kernel 3: seed pooling (4 seeds x 256 keys per partition/head) -> reps.
// Tiny fraction of total FLOPs; plain VALU.
// ---------------------------------------------------------------------------
__global__ __launch_bounds__(256) void pool_kernel(
    const h_t* __restrict__ K16, const h_t* __restrict__ V16,
    const int* __restrict__ pidx, const float* __restrict__ seeds,
    h_t* __restrict__ repsK, h_t* __restrict__ repsV) {
  __shared__ float wbuf[MREPS][SPART];
  __shared__ int nodes[SPART];
  __shared__ float red[SPART];
  const int part = blockIdx.x, h = blockIdx.y;
  const int s = threadIdx.x;
  const int node = pidx[part * SPART + s];
  nodes[s] = node;

  float lg[MREPS] = {0.f, 0.f, 0.f, 0.f};
  const h_t* krow = K16 + (size_t)node * HD + h * DHEAD;
  for (int d = 0; d < DHEAD; ++d) {
    const float kv = (float)krow[d];
#pragma unroll
    for (int mi = 0; mi < MREPS; ++mi)
      lg[mi] += kv * seeds[(mi * NHEADS + h) * DHEAD + d];
  }
#pragma unroll
  for (int mi = 0; mi < MREPS; ++mi) lg[mi] *= SCALE_INV;
  __syncthreads();

  for (int mi = 0; mi < MREPS; ++mi) {
    red[s] = lg[mi]; __syncthreads();
    for (int off = 128; off > 0; off >>= 1) {
      if (s < off) red[s] = fmaxf(red[s], red[s + off]);
      __syncthreads();
    }
    const float mx = red[0]; __syncthreads();
    const float e = __expf(lg[mi] - mx);
    red[s] = e; __syncthreads();
    for (int off = 128; off > 0; off >>= 1) {
      if (s < off) red[s] += red[s + off];
      __syncthreads();
    }
    const float sum = red[0]; __syncthreads();
    wbuf[mi][s] = e / sum;
  }
  __syncthreads();

  for (int p = s; p < MREPS * DHEAD; p += 256) {
    const int mi = p >> 7, d = p & 127;
    float accK = 0.f, accV = 0.f;
    for (int q = 0; q < SPART; ++q) {
      const int nq = nodes[q];
      const float wq_ = wbuf[mi][q];
      accK += wq_ * (float)K16[(size_t)nq * HD + h * DHEAD + d];
      accV += wq_ * (float)V16[(size_t)nq * HD + h * DHEAD + d];
    }
    const int rep = part * MREPS + mi;
    repsK[((size_t)rep * NHEADS + h) * DHEAD + d] = (h_t)accK;
    repsV[((size_t)rep * NHEADS + h) * DHEAD + d] = (h_t)accV;
  }
}

// ---------------------------------------------------------------------------
// Kernel 4: global cross-attention over 512 reps + final combine.
// 1 block = 16 nodes, 4 waves (one per head); combine via LDS, fuse
// x_self (V head-mean) and alpha/beta mixing into the output write.
// ---------------------------------------------------------------------------
__global__ __launch_bounds__(128) void cross_attn_kernel(
    const h_t* __restrict__ Q16, const h_t* __restrict__ V16,
    const h_t* __restrict__ repsK, const h_t* __restrict__ repsV,
    const float* __restrict__ x_local, const float* __restrict__ alpha_logit,
    const float* __restrict__ beta_p, float* __restrict__ out) {
  __shared__ alignas(16) h_t pbuf_s[4][16 * 32];
  __shared__ float ogbuf[4][16][DHEAD];
  const int blk = blockIdx.x;
  const int w = threadIdx.x >> 5;  // head
  const int lane = threadIdx.x & 31;
  const int li = lane & 15, g = lane >> 4;
  const int qnode = blk * 16 + li;

  v16h qfrag[4];
  const h_t* qbase = Q16 + (size_t)qnode * HD + w * DHEAD;
#pragma unroll
  for (int kc = 0; kc < 4; ++kc) qfrag[kc] = load_frag16(qbase + kc * 32, g);
  v8f oacc[8];
#pragma unroll
  for (int t = 0; t < 8; ++t) oacc[t] = (v8f){0.f,0.f,0.f,0.f,0.f,0.f,0.f,0.f};
  float l[8];
  flash_rows_global(repsK, repsV, w, NREPS, qfrag, &pbuf_s[w][0], oacc, l);
#pragma unroll
  for (int t = 0; t < 8; ++t) {
#pragma unroll
    for (int i = 0; i < 8; ++i)
      ogbuf[w][i + 8 * g][t * 16 + li] = oacc[t][i] / l[i];
  }
  __syncthreads();

  const float alpha = 1.f / (1.f + __expf(-alpha_logit[0]));
  const float beta = beta_p[0];
  for (int idx = threadIdx.x; idx < 16 * DHEAD; idx += 128) {
    const int row = idx >> 7, d = idx & 127;
    const int node = blk * 16 + row;
    const float xg = 0.25f * (ogbuf[0][row][d] + ogbuf[1][row][d] +
                              ogbuf[2][row][d] + ogbuf[3][row][d]);
    const float xs =
        0.25f * ((float)V16[(size_t)node * HD + d] +
                 (float)V16[(size_t)node * HD + 128 + d] +
                 (float)V16[(size_t)node * HD + 256 + d] +
                 (float)V16[(size_t)node * HD + 384 + d]);
    out[(size_t)node * DHEAD + d] =
        alpha * x_local[(size_t)node * DHEAD + d] + (1.f - alpha) * xg +
        beta * xs;
  }
}

// ---------------------------------------------------------------------------
extern "C" void kernel_launch(void* const* d_in, const int* in_sizes, int n_in,
                              void* d_out, int out_size, void* d_ws,
                              size_t ws_size, hipStream_t stream) {
  (void)in_sizes; (void)n_in; (void)out_size; (void)ws_size;
  const float* x = (const float*)d_in[0];
  const int* pidx = (const int*)d_in[1];
  const float* wq = (const float*)d_in[2];
  const float* bq = (const float*)d_in[3];
  const float* wk = (const float*)d_in[4];
  const float* bk = (const float*)d_in[5];
  const float* wv = (const float*)d_in[6];
  const float* bv = (const float*)d_in[7];
  const float* seeds = (const float*)d_in[8];
  const float* alog = (const float*)d_in[9];
  const float* beta = (const float*)d_in[10];
  float* out = (float*)d_out;

  // workspace layout (f16 halves first, f32 tail) -- ~115 MB total
  h_t* wt16 = (h_t*)d_ws;                                // 3*512*256
  h_t* Q16 = wt16 + (size_t)3 * HD * IN_CH;              // N*512
  h_t* K16 = Q16 + (size_t)N_NODES * HD;
  h_t* V16 = K16 + (size_t)N_NODES * HD;
  h_t* repsK = V16 + (size_t)N_NODES * HD;               // 512*512
  h_t* repsV = repsK + (size_t)NREPS * HD;
  float* x_local = (float*)(repsV + (size_t)NREPS * HD); // N*128 f32

  convert_w_kernel<<<(3 * IN_CH * HD + 255) / 256, 256, 0, stream>>>(
      wq, wk, wv, wt16);
  qkv_gemm_kernel<<<dim3(N_NODES / 16, HD / 128, 3), 32, 0, stream>>>(
      x, wt16, bq, bk, bv, Q16, K16, V16);
  local_attn_kernel<<<KPART, 512, 0, stream>>>(Q16, K16, V16, pidx, x_local);
  pool_kernel<<<dim3(KPART, NHEADS), 256, 0, stream>>>(K16, V16, pidx, seeds,
                                                       repsK, repsV);
  cross_attn_kernel<<<N_NODES / 16, 128, 0, stream>>>(
      Q16, V16, repsK, repsV, x_local, alog, beta, out);
}
